// Attention_33732673143107
// MI455X (gfx1250) — matrix operand order repaired and checked by reading, not compiled, not run on hardware
//
#include <hip/hip_runtime.h>

// ---------------- problem dims (compile-time constants) ----------------
#define BB   8      // batch
#define CCH  256    // in channels
#define LSEQ 2048   // sequence length
#define DDIM 128    // head dim

// ---------------- fragment types (wave32 WMMA) ----------------
typedef __attribute__((ext_vector_type(16))) _Float16 v16h;
typedef __attribute__((ext_vector_type(8)))  float    v8f;
typedef __attribute__((ext_vector_type(4)))  float    f32x4;
typedef __attribute__((ext_vector_type(4)))  unsigned u32x4;
typedef __attribute__((ext_vector_type(2)))  unsigned u32x2;

union Frag16 {                 // 16 f16 = 8 VGPRs : one WMMA A or B operand
  v16h     h;
  _Float16 e[16];
  unsigned x[8];
  u32x4    u[2];
};

static __device__ __forceinline__ unsigned pk2(float lo, float hi) {
  // v_cvt_pk_rtz_f16_f32
  return __builtin_bit_cast(unsigned int, __builtin_amdgcn_cvt_pkrtz(lo, hi));
}

static __device__ __forceinline__ v8f wmma_f16(v16h a, v16h b, v8f c) {
  // v_wmma_f32_16x16x32_f16
  return __builtin_amdgcn_wmma_f32_16x16x32_f16(false, a, false, b, (short)0, c,
                                                false, false);
}

// log2(e)/sqrt(D): folded into Q so softmax uses raw exp2 (v_exp_f32)
#define QSCALE 0.1275312513582877f

#define WELEM (DDIM * CCH)     // elements per weight matrix (32768)

// =====================================================================
// Kernel 0: one-shot weight conversion f32 -> f16.
// Keeps kernel 1's B fragments as pure b128 loads (no per-tile cvt chains,
// no WMMA->VALU WAR hazard NOPs on the B source registers).
// grid = (WELEM/1024, 1, 3) ; block = 256 ; 4 floats/thread.
// =====================================================================
__launch_bounds__(256, 1)
__global__ void wcvt_kernel(const float* __restrict__ Wq,
                            const float* __restrict__ Wk,
                            const float* __restrict__ Wv,
                            _Float16* __restrict__ w16)
{
  const int z = blockIdx.z;
  const float* __restrict__ src = (z == 0) ? Wq : (z == 1) ? Wk : Wv;
  _Float16* __restrict__ dst = w16 + (size_t)z * WELEM;

  const int i = (blockIdx.x * 256 + threadIdx.x) * 4;
  f32x4 f = *(const f32x4*)(src + i);
  u32x2 p;
  p.x = pk2(f.x, f.y);
  p.y = pk2(f.z, f.w);
  *(u32x2*)(dst + i) = p;
}

// =====================================================================
// Kernel 1: fused QKV projection (f16 weights from workspace).
//   q[b,l,d] = (sum_c x_inner[b,c,l] * Wq[d,c] + bq[d]) * QSCALE -> q_ws [B,L,D] f16
//   k[b,l,d] =  sum_c x_outer[b,c,l] * Wk[d,c] + bk[d]           -> k_ws [B,L,D] f16
//   v       =  ... Wv,bv ...                                     -> v_ws [B,D,L] f16 (transposed!)
// grid = (L/64, B, 2) ; z==0 -> Q, z==1 -> K and V. block = 128 (4 waves).
// =====================================================================
#define ROWPAD 264   // f16 per LDS row: 256 + 8 pad -> 528B rows, 16B aligned, conflict-free reads

__launch_bounds__(128, 1)
__global__ void qkv_proj_kernel(const float* __restrict__ x_inner,
                                const float* __restrict__ x_outer,
                                const _Float16* __restrict__ w16,   // [3][D][C] f16
                                const float* __restrict__ bq,
                                const float* __restrict__ bk,
                                const float* __restrict__ bv,
                                _Float16* __restrict__ q_ws,
                                _Float16* __restrict__ k_ws,
                                _Float16* __restrict__ v_ws)
{
  __shared__ _Float16 xT[64 * ROWPAD];   // x tile transposed: [l][c] in f16 (33.8 KB)

  const int b  = blockIdx.y;
  const int l0 = blockIdx.x * 64;
  const int z  = blockIdx.z;
  const int t  = threadIdx.x;

  const float* __restrict__ x = (z == 0) ? x_inner : x_outer;

  // ---- stage x[b, 0:256, l0:l0+64] into LDS as f16, transposed & c-pair packed ----
  {
    const int p = t;                                   // c-pair: c = 2p, 2p+1
    const float* r0 = x + ((size_t)b * CCH + 2 * p) * LSEQ + l0;
    const float* r1 = r0 + LSEQ;
#pragma unroll
    for (int i = 0; i < 16; ++i) {
      f32x4 a = *(const f32x4*)(r0 + 4 * i);
      f32x4 c = *(const f32x4*)(r1 + 4 * i);
      const int lr = 4 * i;
      *(unsigned*)(xT + (size_t)(lr + 0) * ROWPAD + 2 * p) = pk2(a.x, c.x);
      *(unsigned*)(xT + (size_t)(lr + 1) * ROWPAD + 2 * p) = pk2(a.y, c.y);
      *(unsigned*)(xT + (size_t)(lr + 2) * ROWPAD + 2 * p) = pk2(a.z, c.z);
      *(unsigned*)(xT + (size_t)(lr + 3) * ROWPAD + 2 * p) = pk2(a.w, c.w);
    }
  }
  __syncthreads();

  const int lane = t & 31;
  const int w    = t >> 5;        // wave -> 16 l rows
  const int ln   = lane & 15;
  const int half = lane >> 4;

  // ---- load 8 resident A fragments (16 l x 32 c each) from LDS ----
  // A layout: lane m=ln ; VGPR j: k = (j/4)*16 + half*8 + (j%4)*2
  Frag16 afrag[8];
  {
    const _Float16* arow = xT + (size_t)(w * 16 + ln) * ROWPAD;
#pragma unroll
    for (int ks = 0; ks < 8; ++ks) {
      afrag[ks].u[0] = *(const u32x4*)(arow + ks * 32 + half * 8);
      afrag[ks].u[1] = *(const u32x4*)(arow + ks * 32 + 16 + half * 8);
    }
  }

  auto project = [&](const _Float16* __restrict__ W, const float* __restrict__ bias,
                     float oscale, _Float16* __restrict__ dst, bool transposed) {
#pragma unroll 1
    for (int dt = 0; dt < 8; ++dt) {
      const int d = dt * 16 + ln;
      const float bvv = bias[d];
      v8f acc;
#pragma unroll
      for (int r = 0; r < 8; ++r) acc[r] = bvv;

      const _Float16* wrow = W + (size_t)d * CCH;
#pragma unroll
      for (int ks = 0; ks < 8; ++ks) {
        // B fragment: lane n=d ; VGPR j: k = (j/4)*16 + half*8 + (j%4)*2 (b128 x2)
        Frag16 bf;
        bf.u[0] = *(const u32x4*)(wrow + ks * 32 + half * 8);
        bf.u[1] = *(const u32x4*)(wrow + ks * 32 + 16 + half * 8);
        acc = wmma_f16(afrag[ks].h, bf.h, acc);
      }

      if (!transposed) {
        // D frag: row m = r + half*8, col = d
#pragma unroll
        for (int r = 0; r < 8; ++r) {
          const int row = l0 + w * 16 + half * 8 + r;
          dst[((size_t)b * LSEQ + row) * DDIM + d] = (_Float16)(acc[r] * oscale);
        }
      } else {
        // store V transposed: v_ws[b][d][l] ; 8 consecutive l per lane -> one b128 store
        Frag16 ov;
#pragma unroll
        for (int r = 0; r < 8; ++r) ov.e[r] = (_Float16)(acc[r] * oscale);
        *(u32x4*)(dst + ((size_t)b * DDIM + d) * LSEQ + l0 + w * 16 + half * 8) = ov.u[0];
      }
    }
  };

  if (z == 0) {
    project(w16, bq, QSCALE, q_ws, false);
  } else {
    project(w16 + WELEM,     bk, 1.0f, k_ws, false);
    project(w16 + 2 * WELEM, bv, 1.0f, v_ws, true);
  }
}

// =====================================================================
// Kernel 2: flash attention, one wave per 16-row q tile.
// Computes S^T = K·Q^T so the softmax row index is lane%16:
//   - running max/sum are per-lane scalars
//   - P (f32 D-frag) -> P^T (f16 B-frag) needs only shfl_xor(16) + selects
//   - O^T accumulates with A = V^T (contiguous b128 loads from v_ws)
// grid = (L/128, B), block = 256 (8 waves), no LDS, no inter-wave sync.
// =====================================================================
__launch_bounds__(256, 1)
__global__ void flash_attn_kernel(const _Float16* __restrict__ q_ws,
                                  const _Float16* __restrict__ k_ws,
                                  const _Float16* __restrict__ v_ws,
                                  float* __restrict__ out)
{
  const int b    = blockIdx.y;
  const int t    = threadIdx.x;
  const int lane = t & 31;
  const int w    = t >> 5;
  const int ln   = lane & 15;
  const int half = lane >> 4;
  const bool hiHalf = (half != 0);

  const int l0   = blockIdx.x * 128 + w * 16;
  const int qrow = l0 + ln;

  // ---- resident Q B-fragments (n = q = ln ; k = d) : 4 x 32-deep ----
  Frag16 qf[4];
  {
    const _Float16* qb = q_ws + ((size_t)b * LSEQ + qrow) * DDIM;
#pragma unroll
    for (int ds = 0; ds < 4; ++ds) {
      const u32x4* p = (const u32x4*)(qb + ds * 32 + half * 8);
      qf[ds].u[0] = p[0];
      qf[ds].u[1] = p[2];
    }
  }

  v8f o[8];
#pragma unroll
  for (int dt = 0; dt < 8; ++dt)
#pragma unroll
    for (int r = 0; r < 8; ++r) o[dt][r] = 0.0f;

  float m_run = -3.0e38f;   // running row max (of log2e-scaled scores)
  float s_run = 0.0f;       // running row sum

#pragma unroll 1
  for (int j0 = 0; j0 < LSEQ; j0 += 32) {
    const _Float16* kr0 = k_ws + ((size_t)b * LSEQ + j0 + ln) * DDIM;  // kpos tile 0
    const _Float16* kr1 = kr0 + 16 * DDIM;                             // kpos tile 1

    if (j0 + 32 < LSEQ) {
      __builtin_prefetch(kr0 + 32 * DDIM, 0, 1);                               // next K tile
      __builtin_prefetch(v_ws + ((size_t)b * DDIM + ln) * LSEQ + j0 + 32, 0, 1); // next V tile
    }

    // ---- S^T = K · Q^T  (two 16x16 accum tiles over 32 k-positions) ----
    v8f s0, s1;
#pragma unroll
    for (int r = 0; r < 8; ++r) { s0[r] = 0.0f; s1[r] = 0.0f; }
#pragma unroll
    for (int ds = 0; ds < 4; ++ds) {
      Frag16 ka, kb;
      ka.u[0] = *(const u32x4*)(kr0 + ds * 32 + half * 8);
      ka.u[1] = *(const u32x4*)(kr0 + ds * 32 + 16 + half * 8);
      s0 = wmma_f16(ka.h, qf[ds].h, s0);
      kb.u[0] = *(const u32x4*)(kr1 + ds * 32 + half * 8);
      kb.u[1] = *(const u32x4*)(kr1 + ds * 32 + 16 + half * 8);
      s1 = wmma_f16(kb.h, qf[ds].h, s1);
    }

    // ---- online softmax over this tile's 32 columns (row = q = ln) ----
    float tmax = fmaxf(s0[0], s1[0]);
#pragma unroll
    for (int r = 1; r < 8; ++r) tmax = fmaxf(tmax, fmaxf(s0[r], s1[r]));
    tmax = fmaxf(tmax, __shfl_xor(tmax, 16, 32));       // combine wave halves
    const float mnew = fmaxf(m_run, tmax);
    const float fac  = __builtin_amdgcn_exp2f(m_run - mnew);

    float p0[8], p1[8];
    float ts = 0.0f;
#pragma unroll
    for (int r = 0; r < 8; ++r) {
      p0[r] = __builtin_amdgcn_exp2f(s0[r] - mnew);
      p1[r] = __builtin_amdgcn_exp2f(s1[r] - mnew);
      ts += p0[r] + p1[r];
    }
    ts += __shfl_xor(ts, 16, 32);
    s_run = s_run * fac + ts;
    m_run = mnew;

    // ---- build P^T as a B fragment (k = kpos, n = q) ----
    // value(kpos = h*16 + 2v + s) lives in tile h, reg (2v+s)&7, half (v>=4)
    float c0[8], c1[8];
#pragma unroll
    for (int r = 0; r < 8; ++r) {
      c0[r] = __shfl_xor(p0[r], 16, 32);
      c1[r] = __shfl_xor(p1[r], 16, 32);
    }
    Frag16 pf;
#pragma unroll
    for (int v = 0; v < 8; ++v) {
      const int r = (2 * v) & 7;
      float lo, hi;
      if (v < 4) { lo = hiHalf ? c1[r] : p0[r];  hi = hiHalf ? c1[r + 1] : p0[r + 1]; }
      else       { lo = hiHalf ? p1[r] : c0[r];  hi = hiHalf ? p1[r + 1] : c0[r + 1]; }
      pf.x[v] = pk2(lo, hi);
    }

    // ---- O^T rescale (separate VALU block, then clean WMMA chain) ----
#pragma unroll
    for (int dt = 0; dt < 8; ++dt)
#pragma unroll
      for (int r = 0; r < 8; ++r) o[dt][r] *= fac;

    // ---- O^T += V^T · P^T ----
#pragma unroll
    for (int dt = 0; dt < 8; ++dt) {
      Frag16 va;
      const _Float16* vr = v_ws + ((size_t)b * DDIM + dt * 16 + ln) * LSEQ + j0;
      va.u[0] = *(const u32x4*)(vr + half * 16);
      va.u[1] = *(const u32x4*)(vr + half * 16 + 8);
      o[dt] = wmma_f16(va.h, pf.h, o[dt]);
    }
  }

  // ---- epilogue: out[b][qrow][d] = o^T(d, q) / s_run  (contiguous f32x4 stores) ----
  const float inv = 1.0f / s_run;
  float* ob = out + ((size_t)b * LSEQ + qrow) * DDIM;
#pragma unroll
  for (int dt = 0; dt < 8; ++dt) {
    f32x4 lo4, hi4;
    lo4.x = o[dt][0] * inv; lo4.y = o[dt][1] * inv;
    lo4.z = o[dt][2] * inv; lo4.w = o[dt][3] * inv;
    hi4.x = o[dt][4] * inv; hi4.y = o[dt][5] * inv;
    hi4.z = o[dt][6] * inv; hi4.w = o[dt][7] * inv;
    *(f32x4*)(ob + dt * 16 + half * 8)     = lo4;
    *(f32x4*)(ob + dt * 16 + half * 8 + 4) = hi4;
  }
}

// =====================================================================
extern "C" void kernel_launch(void* const* d_in, const int* in_sizes, int n_in,
                              void* d_out, int out_size, void* d_ws, size_t ws_size,
                              hipStream_t stream) {
  (void)in_sizes; (void)n_in; (void)out_size; (void)ws_size;

  const float* x_inner = (const float*)d_in[0];
  const float* x_outer = (const float*)d_in[1];
  const float* Wq = (const float*)d_in[2];
  const float* bq = (const float*)d_in[3];
  const float* Wk = (const float*)d_in[4];
  const float* bk = (const float*)d_in[5];
  const float* Wv = (const float*)d_in[6];
  const float* bv = (const float*)d_in[7];

  _Float16* q_ws = (_Float16*)d_ws;                       // [B,L,D] f16, 4 MB
  _Float16* k_ws = q_ws + (size_t)BB * LSEQ * DDIM;       // [B,L,D] f16, 4 MB
  _Float16* v_ws = k_ws + (size_t)BB * LSEQ * DDIM;       // [B,D,L] f16, 4 MB
  _Float16* w16  = v_ws + (size_t)BB * LSEQ * DDIM;       // [3][D][C] f16, 192 KB

  dim3 g0(WELEM / 1024, 1, 3);
  wcvt_kernel<<<g0, 256, 0, stream>>>(Wq, Wk, Wv, w16);

  dim3 g1(LSEQ / 64, BB, 2);
  qkv_proj_kernel<<<g1, 128, 0, stream>>>(x_inner, x_outer, w16, bq, bk, bv,
                                          q_ws, k_ws, v_ws);

  dim3 g2(LSEQ / 128, BB);
  flash_attn_kernel<<<g2, 256, 0, stream>>>(q_ws, k_ws, v_ws, (float*)d_out);
}
